// InplaceWorkerTransformerBlock_61649960567208
// MI455X (gfx1250) — matrix operand order repaired and checked by reading, not compiled, not run on hardware
//
#include <hip/hip_runtime.h>
#include <hip/hip_bf16.h>
#include <math.h>

#define DIM   1024
#define HEADS 16
#define HDS   64
#define FFN_D 4096
#define BB    4
#define SS    4096
#define NWK   512
#define EPSF  1.19209290e-07f

typedef unsigned short u16;
typedef __attribute__((ext_vector_type(16))) __bf16 bf16x16;
typedef __attribute__((ext_vector_type(8)))  float  f32x8;
typedef __attribute__((ext_vector_type(8)))  u16    u16x8;

union FragU { u16x8 h[2]; bf16x16 v; };

__device__ inline u16 f2bf(float f) {
  union { float f; unsigned u; } x; x.f = f;
  unsigned r = x.u + 0x7FFFu + ((x.u >> 16) & 1u);
  return (u16)(r >> 16);
}
__device__ inline float bf2f(u16 h) {
  union { unsigned u; float f; } x; x.u = ((unsigned)h) << 16;
  return x.f;
}

// CDNA5 async global->LDS copy (ASYNCcnt-tracked), 16B per lane.
__device__ inline void asyncLoadB128(unsigned ldsOff, const void* g) {
  asm volatile("global_load_async_to_lds_b128 %0, %1, off"
               :: "v"(ldsOff), "v"(g) : "memory");
}
__device__ inline void waitAsync0() {
  asm volatile("s_wait_asynccnt 0x0" ::: "memory");
}
__device__ inline unsigned ldsAddr(const void* p) {
  return (unsigned)(unsigned long long)p;   // LDS aperture: addr[31:0] = offset
}

// ---------------------------------------------------------------------------
// fp32 -> bf16 weight conversion, TRANSPOSED: W[K][N] -> Wt[N][K]
// ---------------------------------------------------------------------------
__global__ __launch_bounds__(256) void cvt_bf16t_kernel(
    const float* __restrict__ src, u16* __restrict__ dst, int K, int N) {
  int i = blockIdx.x * 256 + threadIdx.x;
  if (i < K * N) {
    int k = i / N, n = i - k * N;
    dst[(size_t)n * K + k] = f2bf(src[i]);
  }
}

// concat [w_q | w_gate] transposed -> [2*DIM][DIM] bf16 (row n, col k)
__global__ __launch_bounds__(256) void concat_qg_kernel(
    const float* __restrict__ wq, const float* __restrict__ wg,
    u16* __restrict__ dst) {
  int i = blockIdx.x * 256 + threadIdx.x;  // over DIM*2*DIM
  int k = i >> 11, n = i & 2047;
  dst[(size_t)n * DIM + k] =
      f2bf(n < DIM ? wq[k * DIM + n] : wg[k * DIM + (n - DIM)]);
}

// ---------------------------------------------------------------------------
// block-wide (256 thread / 8 wave32) sum reduction
// ---------------------------------------------------------------------------
__device__ inline float blockSum256(float v) {
#pragma unroll
  for (int off = 1; off < 32; off <<= 1) v += __shfl_xor(v, off, 32);
  __shared__ float red[8];
  int lane = threadIdx.x & 31, wv = threadIdx.x >> 5;
  if (lane == 0) red[wv] = v;
  __syncthreads();
  float s = 0.0f;
#pragma unroll
  for (int i = 0; i < 8; ++i) s += red[i];
  return s;
}

// conv mix (width-3 causal) + RMSNorm -> h bf16  (one block per token)
__global__ __launch_bounds__(256) void conv_rms_kernel(
    const float* __restrict__ x, const float* __restrict__ cw,
    const float* __restrict__ cb, const float* __restrict__ lnw,
    u16* __restrict__ h) {
  int row = blockIdx.x;            // b*SS + s
  int s = row & (SS - 1);
  int c = threadIdx.x * 4;
  const float* xr = x + (size_t)row * DIM;
  float mx[4]; float ss = 0.0f;
#pragma unroll
  for (int e = 0; e < 4; ++e) {
    int ci = c + e;
    float v = xr[ci] * cw[2 * DIM + ci] + cb[ci];
    if (s >= 1) v += x[((size_t)row - 1) * DIM + ci] * cw[DIM + ci];
    if (s >= 2) v += x[((size_t)row - 2) * DIM + ci] * cw[ci];
    mx[e] = v; ss += v * v;
  }
  float tot = blockSum256(ss);
  float sc = rsqrtf(tot * (1.0f / DIM) + EPSF);
#pragma unroll
  for (int e = 0; e < 4; ++e)
    h[(size_t)row * DIM + c + e] = f2bf(mx[e] * sc * lnw[c + e]);
}

// plain RMSNorm (worker rows) -> bf16
__global__ __launch_bounds__(256) void rms2_kernel(
    const float* __restrict__ wx, const float* __restrict__ lnw,
    u16* __restrict__ h2) {
  int row = blockIdx.x;
  int c = threadIdx.x * 4;
  const float* xr = wx + (size_t)row * DIM;
  float v[4]; float ss = 0.0f;
#pragma unroll
  for (int e = 0; e < 4; ++e) { v[e] = xr[c + e]; ss += v[e] * v[e]; }
  float tot = blockSum256(ss);
  float sc = rsqrtf(tot * (1.0f / DIM) + EPSF);
#pragma unroll
  for (int e = 0; e < 4; ++e)
    h2[(size_t)row * DIM + c + e] = f2bf(v[e] * sc * lnw[c + e]);
}

// ---------------------------------------------------------------------------
// WMMA bf16 GEMM: C[M,N] = A[M,K] * Bt[N,K]^T, 128x128x32 tiles, 256 threads.
// A and B tiles staged with global_load_async_to_lds_b128, double-buffered;
// one s_wait_asynccnt + one barrier per k-step. Frag reads are b128 ds loads.
// Epilogues:
//  MODE 0: bf16 store
//  MODE 1: +bias, exact GELU, bf16 store            (ffn1)
//  MODE 3: + x[b*SS+widx[w]] gather, f32 store      (attn_out -> worker_x)
//  MODE 2: +bias, + aux row, scatter to out rows    (ffn2 -> d_out)
// ---------------------------------------------------------------------------
template <int MODE>
__global__ __launch_bounds__(256) void gemm_bf16_kernel(
    const u16* __restrict__ A, const u16* __restrict__ BT, int M, int N, int K,
    void* __restrict__ C, int ldC, const float* __restrict__ bias,
    const float* __restrict__ aux, const int* __restrict__ widx) {
  __shared__ __align__(16) u16 As[2][128 * 32];
  __shared__ __align__(16) u16 Bs[2][128 * 32];   // [n][k]
  const int tid = threadIdx.x;
  const int lane = tid & 31, wave = tid >> 5;
  const int wm = wave & 3, wn = wave >> 2;        // 4 x 2 wave grid
  const int m0 = blockIdx.y * 128, n0 = blockIdx.x * 128;
  const int r0 = lane & 15, half = lane >> 4;
  const int akb = half ? 8 : 0, bkb = half ? 8 : 0;

  f32x8 acc[2][4];
#pragma unroll
  for (int i = 0; i < 2; ++i)
#pragma unroll
    for (int j = 0; j < 4; ++j)
#pragma unroll
      for (int e = 0; e < 8; ++e) acc[i][j][e] = 0.0f;

  // async tile fetch: 512 x 16B chunks per matrix, 2 per thread
  auto issueTile = [&](int buf, int k0) {
#pragma unroll
    for (int it = 0; it < 2; ++it) {
      int i = tid + it * 256;
      int r = i >> 2, c8 = (i & 3) * 8;
      asyncLoadB128(ldsAddr(&As[buf][r * 32 + c8]),
                    &A[(size_t)(m0 + r) * K + k0 + c8]);
      asyncLoadB128(ldsAddr(&Bs[buf][r * 32 + c8]),
                    &BT[(size_t)(n0 + r) * K + k0 + c8]);
    }
  };

  const int nk = K >> 5;
  issueTile(0, 0);
  for (int kt = 0; kt < nk; ++kt) {
    waitAsync0();
    __syncthreads();
    if (kt + 1 < nk) issueTile((kt + 1) & 1, (kt + 1) << 5);
    const u16* as = As[kt & 1];
    const u16* bs = Bs[kt & 1];
    bf16x16 af[2], bfr[4];
#pragma unroll
    for (int i = 0; i < 2; ++i) {
      int r = wm * 32 + i * 16 + r0;
      FragU t;
      t.h[0] = *(const u16x8*)&as[r * 32 + akb];
      t.h[1] = *(const u16x8*)&as[r * 32 + akb + 16];
      af[i] = t.v;
    }
#pragma unroll
    for (int j = 0; j < 4; ++j) {
      int c = wn * 64 + j * 16 + r0;
      FragU t;
      t.h[0] = *(const u16x8*)&bs[c * 32 + bkb];
      t.h[1] = *(const u16x8*)&bs[c * 32 + bkb + 16];
      bfr[j] = t.v;
    }
#pragma unroll
    for (int i = 0; i < 2; ++i)
#pragma unroll
      for (int j = 0; j < 4; ++j)
        acc[i][j] = __builtin_amdgcn_wmma_f32_16x16x32_bf16(
            false, af[i], false, bfr[j], (short)0, acc[i][j], false, false);
  }

#pragma unroll
  for (int i = 0; i < 2; ++i)
#pragma unroll
    for (int j = 0; j < 4; ++j) {
      int gmBase = m0 + wm * 32 + i * 16 + half * 8;
      int gn = n0 + wn * 64 + j * 16 + r0;
#pragma unroll
      for (int e = 0; e < 8; ++e) {
        int m = gmBase + e;
        float v = acc[i][j][e];
        if (MODE == 0) {
          ((u16*)C)[(size_t)m * ldC + gn] = f2bf(v);
        } else if (MODE == 1) {
          v += bias[gn];
          float g = 0.5f * v * (1.0f + erff(v * 0.70710678118f));
          ((u16*)C)[(size_t)m * ldC + gn] = f2bf(g);
        } else if (MODE == 3) {
          int b = m >> 9, w = m & (NWK - 1);
          float r = v + aux[((size_t)b * SS + widx[w]) * DIM + gn];
          ((float*)C)[(size_t)m * ldC + gn] = r;
        } else {  // MODE 2
          v += bias[gn];
          int b = m >> 9, w = m & (NWK - 1);
          float r = aux[(size_t)m * DIM + gn] + v;
          ((float*)C)[((size_t)b * SS + widx[w]) * ldC + gn] = r;
        }
      }
    }
}

// ---------------------------------------------------------------------------
// RoPE + reshape: kv[M,2048] bf16 -> K [B,H,S,64] (roped), V^T [B,H,64,S]
// ---------------------------------------------------------------------------
__global__ __launch_bounds__(64) void rope_kv_kernel(
    const u16* __restrict__ kv, const float* __restrict__ sinp,
    const float* __restrict__ cosp, u16* __restrict__ kout,
    u16* __restrict__ vT) {
  int d = threadIdx.x;
  int idx = blockIdx.x;            // b*HEADS*SS + h*SS + s
  int s = idx & (SS - 1);
  int bh = idx >> 12;
  int hh = bh & (HEADS - 1);
  int b = bh >> 4;
  size_t row = (size_t)b * SS + s;
  int pair = (d < 32) ? d + 32 : d - 32;
  float kd = bf2f(kv[row * (2 * DIM) + hh * HDS + d]);
  float kp = bf2f(kv[row * (2 * DIM) + hh * HDS + pair]);
  float sn = sinp[row * HDS + d];
  float cn = cosp[row * HDS + d];
  float kr = kd * cn + ((d < 32) ? -kp : kp) * sn;
  kout[((size_t)(b * HEADS + hh) * SS + s) * HDS + d] = f2bf(kr);
  float vd = bf2f(kv[row * (2 * DIM) + DIM + hh * HDS + d]);
  vT[((size_t)(b * HEADS + hh) * HDS + d) * SS + s] = f2bf(vd);
}

// RoPE on worker queries: qg[M,2048] cols 0..1023 -> Q [B,H,NW,64]
__global__ __launch_bounds__(64) void rope_q_kernel(
    const u16* __restrict__ qg, const float* __restrict__ sinp,
    const float* __restrict__ cosp, const int* __restrict__ widx,
    u16* __restrict__ qout) {
  int d = threadIdx.x;
  int idx = blockIdx.x;            // b*HEADS*NW + h*NW + w
  int w = idx & (NWK - 1);
  int bh = idx >> 9;
  int hh = bh & (HEADS - 1);
  int b = bh >> 4;
  size_t row = (size_t)b * NWK + w;
  int pair = (d < 32) ? d + 32 : d - 32;
  float qd = bf2f(qg[row * (2 * DIM) + hh * HDS + d]);
  float qp = bf2f(qg[row * (2 * DIM) + hh * HDS + pair]);
  int sidx = widx[w];
  float sn = sinp[(size_t)sidx * HDS + d];   // sin[0, worker_idx]
  float cn = cosp[(size_t)sidx * HDS + d];
  qout[((size_t)(b * HEADS + hh) * NWK + w) * HDS + d] =
      f2bf(qd * cn + ((d < 32) ? -qp : qp) * sn);
}

// gather h rows at worker positions -> wh bf16 [B*NW, DIM]
__global__ __launch_bounds__(256) void gather_wh_kernel(
    const u16* __restrict__ h, const int* __restrict__ widx,
    u16* __restrict__ wh) {
  int row = blockIdx.x;            // b*NW + w
  int b = row >> 9, w = row & (NWK - 1);
  size_t src = ((size_t)b * SS + widx[w]) * DIM;
  const uint2* sp = (const uint2*)(h + src);
  uint2* dp = (uint2*)(wh + (size_t)row * DIM);
  dp[threadIdx.x] = sp[threadIdx.x];
}

// ctx * sigmoid(gate) -> bf16
__global__ __launch_bounds__(256) void ctx_gate_kernel(
    const float* __restrict__ ctx, const u16* __restrict__ qg,
    u16* __restrict__ ctxg) {
  int i = blockIdx.x * 256 + threadIdx.x;    // < B*NW*DIM
  int row = i >> 10, col = i & (DIM - 1);
  float g = bf2f(qg[(size_t)row * (2 * DIM) + DIM + col]);
  float sig = 1.0f / (1.0f + __expf(-g));
  ctxg[i] = f2bf(ctx[i] * sig);
}

// ---------------------------------------------------------------------------
// Flash attention: 1 block = (b, h, 64 q-rows), 4 wave32, 16 q-rows/wave.
// Streams 32-key chunks up to the block's max worker_idx; K/V chunks staged
// via async global->LDS copies.
// ---------------------------------------------------------------------------
__global__ __launch_bounds__(128) void attn_kernel(
    const u16* __restrict__ Q, const u16* __restrict__ Kt,
    const u16* __restrict__ Vt, const int* __restrict__ widx,
    float* __restrict__ ctx) {
  __shared__ __align__(16) u16 Ks[32 * 64];       // [s][d]
  __shared__ __align__(16) u16 Vs[64 * 32];       // [d][s]
  __shared__ __align__(16) u16 Ps[4][16 * 32];    // per-wave probs
  int blk = blockIdx.x;
  int qb = blk & 7;                // NW/64 = 8 q-blocks
  int bh = blk >> 3;
  int hh = bh & (HEADS - 1);
  int b = bh >> 4;
  int tid = threadIdx.x, lane = tid & 31, wave = tid >> 5;
  int qw = qb * 64 + wave * 16;    // first worker row for this wave
  const u16* Qbase = Q + ((size_t)(b * HEADS + hh) * NWK + qw) * HDS;
  const u16* Kbase = Kt + (size_t)(b * HEADS + hh) * SS * HDS;
  const u16* Vbase = Vt + (size_t)(b * HEADS + hh) * HDS * SS;
  int r0 = lane & 15, half = lane >> 4;
  int akb = half ? 8 : 0, bkb = half ? 8 : 0;

  bf16x16 qf[2];
#pragma unroll
  for (int dc = 0; dc < 2; ++dc) {
    FragU t;
    t.h[0] = *(const u16x8*)&Qbase[r0 * HDS + dc * 32 + akb];
    t.h[1] = *(const u16x8*)&Qbase[r0 * HDS + dc * 32 + akb + 16];
    qf[dc] = t.v;
  }
  int wi[8];
#pragma unroll
  for (int j = 0; j < 8; ++j) wi[j] = widx[qw + j + 8 * half];
  float mrow[8], lrow[8];
  f32x8 cacc[4];
#pragma unroll
  for (int j = 0; j < 8; ++j) { mrow[j] = -3.0e38f; lrow[j] = 0.0f; }
#pragma unroll
  for (int t = 0; t < 4; ++t)
#pragma unroll
    for (int e = 0; e < 8; ++e) cacc[t][e] = 0.0f;

  int s_end = widx[qb * 64 + 63] + 1;      // causal bound for this block
  int nchunk = (s_end + 31) >> 5;
  for (int ch = 0; ch < nchunk; ++ch) {
    int s0 = ch << 5;
#pragma unroll
    for (int it = 0; it < 2; ++it) {       // K chunk 32x64 (async)
      int i = tid + it * 128;
      int sr = i >> 3, c8 = (i & 7) * 8;
      asyncLoadB128(ldsAddr(&Ks[sr * 64 + c8]),
                    &Kbase[(size_t)(s0 + sr) * HDS + c8]);
    }
#pragma unroll
    for (int it = 0; it < 2; ++it) {       // V^T chunk 64x32 (async)
      int i = tid + it * 128;
      int dr = i >> 2, c8 = (i & 3) * 8;
      asyncLoadB128(ldsAddr(&Vs[dr * 32 + c8]),
                    &Vbase[(size_t)dr * SS + s0 + c8]);
    }
    waitAsync0();
    __syncthreads();

    f32x8 sc[2];
#pragma unroll
    for (int kt = 0; kt < 2; ++kt)
#pragma unroll
      for (int e = 0; e < 8; ++e) sc[kt][e] = 0.0f;
#pragma unroll
    for (int kt = 0; kt < 2; ++kt)
#pragma unroll
      for (int dc = 0; dc < 2; ++dc) {
        FragU t;   // B frag: col = key slot, rows = d -> contiguous K row
        t.h[0] = *(const u16x8*)&Ks[(kt * 16 + r0) * 64 + dc * 32 + bkb];
        t.h[1] = *(const u16x8*)&Ks[(kt * 16 + r0) * 64 + dc * 32 + bkb + 16];
        sc[kt] = __builtin_amdgcn_wmma_f32_16x16x32_bf16(
            false, qf[dc], false, t.v, (short)0, sc[kt], false, false);
      }

#pragma unroll
    for (int j = 0; j < 8; ++j) {
      float s0v = sc[0][j] * 0.125f;       // 1/sqrt(64)
      float s1v = sc[1][j] * 0.125f;
      int w = wi[j];
      if (s0 + r0 > w) s0v = -3.0e38f;
      if (s0 + 16 + r0 > w) s1v = -3.0e38f;
      float rmax = fmaxf(s0v, s1v);
#pragma unroll
      for (int off = 1; off < 16; off <<= 1)
        rmax = fmaxf(rmax, __shfl_xor(rmax, off, 16));
      float mnew = fmaxf(mrow[j], rmax);
      float alpha = __expf(mrow[j] - mnew);
      float p0 = __expf(s0v - mnew);
      float p1 = __expf(s1v - mnew);
      float rs = p0 + p1;
#pragma unroll
      for (int off = 1; off < 16; off <<= 1) rs += __shfl_xor(rs, off, 16);
      lrow[j] = lrow[j] * alpha + rs;
      mrow[j] = mnew;
#pragma unroll
      for (int t = 0; t < 4; ++t) cacc[t][j] *= alpha;
      Ps[wave][(j + 8 * half) * 32 + r0] = f2bf(p0);
      Ps[wave][(j + 8 * half) * 32 + 16 + r0] = f2bf(p1);
    }
    __syncthreads();

    FragU pa;   // A frag over P (16 q x 32 keys)
    pa.h[0] = *(const u16x8*)&Ps[wave][r0 * 32 + akb];
    pa.h[1] = *(const u16x8*)&Ps[wave][r0 * 32 + akb + 16];
#pragma unroll
    for (int t = 0; t < 4; ++t) {
      FragU vb;  // B frag: col = d, rows = key -> contiguous V^T row
      vb.h[0] = *(const u16x8*)&Vs[(t * 16 + r0) * 32 + bkb];
      vb.h[1] = *(const u16x8*)&Vs[(t * 16 + r0) * 32 + bkb + 16];
      cacc[t] = __builtin_amdgcn_wmma_f32_16x16x32_bf16(
          false, pa.v, false, vb.v, (short)0, cacc[t], false, false);
    }
    __syncthreads();
  }

#pragma unroll
  for (int t = 0; t < 4; ++t)
#pragma unroll
    for (int j = 0; j < 8; ++j) {
      int m = qw + j + 8 * half;
      ctx[(size_t)(b * NWK + m) * DIM + hh * HDS + t * 16 + r0] =
          cacc[t][j] / lrow[j];
    }
}

// ---------------------------------------------------------------------------
extern "C" void kernel_launch(void* const* d_in, const int* in_sizes, int n_in,
                              void* d_out, int out_size, void* d_ws,
                              size_t ws_size, hipStream_t stream) {
  (void)in_sizes; (void)n_in; (void)out_size; (void)ws_size;
  const float* x    = (const float*)d_in[0];
  const float* sinp = (const float*)d_in[1];
  const float* cosp = (const float*)d_in[2];
  const int*   widx = (const int*)d_in[3];
  const float* cw   = (const float*)d_in[4];
  const float* cb   = (const float*)d_in[5];
  const float* wkv  = (const float*)d_in[6];
  const float* wq   = (const float*)d_in[7];
  const float* wg   = (const float*)d_in[8];
  const float* wo   = (const float*)d_in[9];
  const float* ln1  = (const float*)d_in[10];
  const float* ln2  = (const float*)d_in[11];
  const float* fw1  = (const float*)d_in[12];
  const float* fb1  = (const float*)d_in[13];
  const float* fw2  = (const float*)d_in[14];
  const float* fb2  = (const float*)d_in[15];
  float* out = (float*)d_out;

  char* base = (char*)d_ws; size_t off = 0;
  auto carve = [&](size_t bytes) -> void* {
    void* p = base + off; off += (bytes + 255) & ~(size_t)255; return p;
  };
  u16* wkv_b = (u16*)carve((size_t)DIM * 2 * DIM * 2);   // [2048][1024] (N,K)
  u16* wqg_b = (u16*)carve((size_t)DIM * 2 * DIM * 2);   // [2048][1024]
  u16* wo_b  = (u16*)carve((size_t)DIM * DIM * 2);       // [1024][1024]
  u16* w1_b  = (u16*)carve((size_t)DIM * FFN_D * 2);     // [4096][1024]
  u16* w2_b  = (u16*)carve((size_t)FFN_D * DIM * 2);     // [1024][4096]
  u16* h_b   = (u16*)carve((size_t)BB * SS * DIM * 2);
  u16* kv_b  = (u16*)carve((size_t)BB * SS * 2 * DIM * 2);
  u16* k_b   = (u16*)carve((size_t)BB * HEADS * SS * HDS * 2);
  u16* vT_b  = (u16*)carve((size_t)BB * HEADS * HDS * SS * 2);
  u16* wh_b  = (u16*)carve((size_t)BB * NWK * DIM * 2);
  u16* qg_b  = (u16*)carve((size_t)BB * NWK * 2 * DIM * 2);
  u16* q_b   = (u16*)carve((size_t)BB * HEADS * NWK * HDS * 2);
  float* ctx_f  = (float*)carve((size_t)BB * NWK * DIM * 4);
  u16*   ctxg_b = (u16*)carve((size_t)BB * NWK * DIM * 2);
  float* wx_f   = (float*)carve((size_t)BB * NWK * DIM * 4);
  u16*   h2_b   = (u16*)carve((size_t)BB * NWK * DIM * 2);
  u16*   mid_b  = (u16*)carve((size_t)BB * NWK * FFN_D * 2);

  // pass-through copy of x (non-worker rows)
  hipMemcpyAsync(d_out, x, (size_t)BB * SS * DIM * sizeof(float),
                 hipMemcpyDeviceToDevice, stream);

  // weight conversion (transposed to [N][K] bf16 for async B-tile staging)
  int nkv = DIM * 2 * DIM;
  cvt_bf16t_kernel<<<(nkv + 255) / 256, 256, 0, stream>>>(wkv, wkv_b, DIM,
                                                          2 * DIM);
  concat_qg_kernel<<<(nkv + 255) / 256, 256, 0, stream>>>(wq, wg, wqg_b);
  cvt_bf16t_kernel<<<(DIM * DIM + 255) / 256, 256, 0, stream>>>(wo, wo_b, DIM,
                                                                DIM);
  cvt_bf16t_kernel<<<(DIM * FFN_D + 255) / 256, 256, 0, stream>>>(
      fw1, w1_b, DIM, FFN_D);
  cvt_bf16t_kernel<<<(FFN_D * DIM + 255) / 256, 256, 0, stream>>>(
      fw2, w2_b, FFN_D, DIM);

  // conv mix + rmsnorm -> h
  conv_rms_kernel<<<BB * SS, 256, 0, stream>>>(x, cw, cb, ln1, h_b);

  // kv = h @ w_kv   (16384 x 2048 x 1024)
  gemm_bf16_kernel<0><<<dim3((2 * DIM) / 128, (BB * SS) / 128), 256, 0,
                        stream>>>(h_b, wkv_b, BB * SS, 2 * DIM, DIM, kv_b,
                                  2 * DIM, nullptr, nullptr, nullptr);

  rope_kv_kernel<<<BB * HEADS * SS, 64, 0, stream>>>(kv_b, sinp, cosp, k_b,
                                                     vT_b);

  gather_wh_kernel<<<BB * NWK, 256, 0, stream>>>(h_b, widx, wh_b);

  // [q | gate] = worker_h @ [w_q | w_gate]   (2048 x 2048 x 1024)
  gemm_bf16_kernel<0><<<dim3((2 * DIM) / 128, (BB * NWK) / 128), 256, 0,
                        stream>>>(wh_b, wqg_b, BB * NWK, 2 * DIM, DIM, qg_b,
                                  2 * DIM, nullptr, nullptr, nullptr);

  rope_q_kernel<<<BB * HEADS * NWK, 64, 0, stream>>>(qg_b, sinp, cosp, widx,
                                                     q_b);

  attn_kernel<<<BB * HEADS * (NWK / 64), 128, 0, stream>>>(q_b, k_b, vT_b,
                                                           widx, ctx_f);

  ctx_gate_kernel<<<(BB * NWK * DIM) / 256, 256, 0, stream>>>(ctx_f, qg_b,
                                                              ctxg_b);

  // worker_x = x[worker] + (ctx*gate) @ w_out
  gemm_bf16_kernel<3><<<dim3(DIM / 128, (BB * NWK) / 128), 256, 0, stream>>>(
      ctxg_b, wo_b, BB * NWK, DIM, DIM, wx_f, DIM, nullptr, x, widx);

  rms2_kernel<<<BB * NWK, 256, 0, stream>>>(wx_f, ln2, h2_b);

  // mid = gelu(h2 @ ffn_w1 + b1)
  gemm_bf16_kernel<1><<<dim3(FFN_D / 128, (BB * NWK) / 128), 256, 0, stream>>>(
      h2_b, w1_b, BB * NWK, FFN_D, DIM, mid_b, FFN_D, fb1, nullptr, nullptr);

  // out[worker rows] = worker_x + mid @ ffn_w2 + b2
  gemm_bf16_kernel<2><<<dim3(DIM / 128, (BB * NWK) / 128), 256, 0, stream>>>(
      mid_b, w2_b, BB * NWK, DIM, FFN_D, out, DIM, fb2, wx_f, widx);
}
